// CGTPEL_72645076844777
// MI455X (gfx1250) — matrix operand order repaired and checked by reading, not compiled, not run on hardware
//
#include <hip/hip_runtime.h>
#include <hip/hip_bf16.h>

// ---------------------------------------------------------------------------
// Equivariant conv layer, fused for MI455X (gfx1250, wave32, WMMA).
// One fused kernel runs FC1 (relu MLP) entirely in LDS, then FC2
// (E x 128 @ 128 x 4096, the 63-GFLOP hot spot) through f16 WMMA with each
// 16x16 w-tile consumed in-register by the tensor product -> the ~1 GB
// per-edge weight tensor never touches HBM, and h never leaves LDS.
// 32 edges/block: every W2 B-fragment load feeds two WMMAs (halves L2 BW).
// edge_attr tiles are staged with GLOBAL_LOAD_ASYNC_TO_LDS when available.
// ---------------------------------------------------------------------------

typedef _Float16 v16h __attribute__((ext_vector_type(16)));
typedef _Float16 v8h  __attribute__((ext_vector_type(8)));
typedef _Float16 v4h  __attribute__((ext_vector_type(4)));
typedef float    v8f  __attribute__((ext_vector_type(8)));
typedef int      v4i32 __attribute__((vector_size(16)));

#define MUL       32
#define IN_DIM    128
#define HIDDEN    128
#define W_NUMEL   4096
#define EPS_BN    1e-5f
#define INV_SQRT3 0.57735026918962576451f
#define PATH_NORM 0.125f   /* 1/sqrt(2*MUL) = 1/8 */

#if defined(__gfx1250__) && __has_builtin(__builtin_amdgcn_global_load_async_to_lds_b128)
#define HAVE_ASYNC_LDS 1
#define AS_GLOBAL __attribute__((address_space(1)))
#define AS_SHARED __attribute__((address_space(3)))
#endif

// ---------------------------------------------------------------------------
__global__ void ws_zero_kernel(float* __restrict__ p, int n) {
    int i = blockIdx.x * blockDim.x + threadIdx.x;
    if (i < n) p[i] = 0.0f;
}

// ---------------------------------------------------------------------------
// Pack a KxNcols f32 matrix into f16 WMMA B-fragment order:
//   p = ((t*KB + kb)*32 + lane)*16 + j
//   lane<16 holds K = kb*32 + {0..7,16..23}[j], lane>=16 holds K = kb*32 + {8..15,24..31}[j],
//   column N = t*16 + (lane&15).   (16x16x32 f16 B layout, ISA 7.12.2)
// Each lane then loads its whole fragment as 32 contiguous bytes.
// ---------------------------------------------------------------------------
__global__ void pack_b_kernel(const float* __restrict__ src, _Float16* __restrict__ dst,
                              int K, int Ncols, int total) {
    int p = blockIdx.x * blockDim.x + threadIdx.x;
    if (p >= total) return;
    int j    = p & 15;
    int lane = (p >> 4) & 31;
    int frag = p >> 9;
    int KB   = K >> 5;
    int kb   = frag % KB;
    int t    = frag / KB;
    int Kidx = kb * 32 + ((lane & 16) ? 8 : 0) + (j < 8 ? j : j + 8);
    int Nidx = t * 16 + (lane & 15);
    dst[p] = (_Float16)src[(size_t)Kidx * Ncols + Nidx];
}

// A-fragment (16x32 f16) loader from an LDS row-major tile (row stride 128):
// lanes 0-15: row=lane,    K = {0..7, 16..23} of the 32-block
// lanes16-31: row=lane-16, K = {8..15, 24..31}
__device__ __forceinline__ v16h load_a_frag(const _Float16* __restrict__ lds,
                                            int lane, int kb) {
    const _Float16* ap = lds + (lane & 15) * 128 + kb * 32 + ((lane & 16) ? 8 : 0);
    v8h lo = *(const v8h*)(ap);
    v8h hi = *(const v8h*)(ap + 16);
    return __builtin_shufflevector(lo, hi, 0,1,2,3,4,5,6,7,8,9,10,11,12,13,14,15);
}

// ---------------------------------------------------------------------------
// Fused FC1 + FC2 + tensor product + scatter.  32 edges/block, 8 waves.
//
// LDS (49.5 KB, aliased):
//   [0,8K)    x-tile (edge_attr f16)
//   [8K,24K)  f32 async staging for edge_attr (dead after conversion)
//   [8K,16K)  h-tile (relu(fc1) f16, overwrites staging)
//   [0,24K)   accumulators accA/B/D/C0/C1/C2 (after A-fragment barrier)
//   [24K,48K) coefA/B/D/C0/C1/C2 (per-edge TP coefficients)
//   [48K,..)  shv (edge_sh[1:4] per edge)
//
// Wave wv owns FC2 weight block X = wv/2 (A,B,C,D) and output parity wv&1.
// Tile t = X*64 + parity + 2*u covers w[e, X, u, parity*16 .. +15]; each
// B-fragment load is reused by the two 16-edge A-tiles (r = 0,1).
// ---------------------------------------------------------------------------
__global__ __launch_bounds__(256) void fused_edge_kernel(
        const float* __restrict__ edge_attr,
        const _Float16* __restrict__ w1p, const float* __restrict__ b1,
        const _Float16* __restrict__ w2p, const float* __restrict__ b2,
        const float* __restrict__ node_attr, const int* __restrict__ eidx,
        const float* __restrict__ edge_sh,
        float* __restrict__ node_acc, float* __restrict__ counts,
        int E, int N) {
    __shared__ __align__(32) char smem[49536];
    _Float16* x_lds = (_Float16*)(smem);              // 32x128 f16
    _Float16* h_lds = (_Float16*)(smem + 8192);       // 32x128 f16
    float* stage = (float*)(smem + 8192);             // 32x128 f32 staging
    float* accb = (float*)(smem);                     // 6 x 1024 f32 (aliases x/h)
    float* coef = (float*)(smem + 24576);             // 6 x 1024 f32
    float* shv  = (float*)(smem + 49152);             // 32 x 3 f32
    float* coefA  = coef;        float* coefB  = coef + 1024;
    float* coefD  = coef + 2048; float* coefC0 = coef + 3072;
    float* coefC1 = coef + 4096; float* coefC2 = coef + 5120;
    float* accA   = accb;        float* accB   = accb + 1024;
    float* accD   = accb + 2048; float* accC0  = accb + 3072;
    float* accC1  = accb + 4096; float* accC2  = accb + 5120;

    const int tid  = threadIdx.x;
    const int e0   = blockIdx.x * 32;
    const int lane = tid & 31, wv = tid >> 5;
    const int n = lane & 15, mbase = (lane & 16) ? 8 : 0;

    // ---- phase 0: stage edge_attr tile; overlap TP-coefficient gather ----
#ifdef HAVE_ASYNC_LDS
    if (e0 + 32 <= E) {
        // per-lane async b128 copies: 1024 x float4 = 16 KB, ASYNCcnt-tracked
        const float4* src = (const float4*)(edge_attr + (size_t)e0 * 128);
        for (int q = tid; q < 1024; q += 256) {
            __builtin_amdgcn_global_load_async_to_lds_b128(
                (AS_GLOBAL v4i32*)const_cast<float4*>(src + q),
                (AS_SHARED v4i32*)(stage + q * 4),
                0, 0);
        }
    } else {
        for (int p = tid; p < 4096; p += 256) {
            int e = e0 + (p >> 7);
            stage[p] = (e < E) ? edge_attr[(size_t)e * 128 + (p & 127)] : 0.0f;
        }
    }
#endif
    // per-edge tensor-product coefficients (gather node_attr[edge_dst]);
    // runs while the async copies are in flight
    for (int p = tid; p < 1024; p += 256) {
        int m = p >> 5, u = p & 31;
        int e = e0 + m;
        float sh0 = 0.f, s1 = 0.f, s2 = 0.f, s3 = 0.f;
        float x0 = 0.f, xa = 0.f, xb = 0.f, xc = 0.f;
        if (e < E) {
            int dst = eidx[E + e];                        // edge_index[1][e]
            sh0 = edge_sh[(size_t)e * 4 + 0];
            s1  = edge_sh[(size_t)e * 4 + 1];
            s2  = edge_sh[(size_t)e * 4 + 2];
            s3  = edge_sh[(size_t)e * 4 + 3];
            const float* xr = node_attr + (size_t)dst * 128;
            x0 = xr[u];
            xa = xr[32 + u * 3 + 0];
            xb = xr[32 + u * 3 + 1];
            xc = xr[32 + u * 3 + 2];
        }
        coefA[p]  = x0 * sh0;
        coefB[p]  = x0;
        coefC0[p] = xa * sh0;
        coefC1[p] = xb * sh0;
        coefC2[p] = xc * sh0;
        coefD[p]  = xa * s1 + xb * s2 + xc * s3;
        if (u == 0) { shv[m * 3 + 0] = s1; shv[m * 3 + 1] = s2; shv[m * 3 + 2] = s3; }
    }

#ifdef HAVE_ASYNC_LDS
    #if __has_builtin(__builtin_amdgcn_s_wait_asynccnt)
    __builtin_amdgcn_s_wait_asynccnt(0);
    #else
    asm volatile("s_wait_asynccnt 0x0" ::: "memory");
    #endif
    __syncthreads();
    // f32 staging -> f16 x-tile
    for (int q = tid; q < 1024; q += 256) {
        float4 f = ((const float4*)stage)[q];
        ((v4h*)x_lds)[q] = (v4h){(_Float16)f.x, (_Float16)f.y,
                                 (_Float16)f.z, (_Float16)f.w};
    }
#else
    if (e0 + 32 <= E) {
        const float4* src = (const float4*)(edge_attr + (size_t)e0 * 128);
        for (int q = tid; q < 1024; q += 256) {
            float4 f = src[q];
            ((v4h*)x_lds)[q] = (v4h){(_Float16)f.x, (_Float16)f.y,
                                     (_Float16)f.z, (_Float16)f.w};
        }
    } else {
        for (int p = tid; p < 4096; p += 256) {
            int e = e0 + (p >> 7);
            x_lds[p] = (e < E) ? (_Float16)edge_attr[(size_t)e * 128 + (p & 127)]
                               : (_Float16)0.0f;
        }
    }
#endif
    __syncthreads();

    // ---- phase 1: FC1, h = relu(x @ W1 + b1), stays in LDS ---------------
    {
        const int ct = wv;                     // hidden column tile 0..7
        float bval = b1[ct * 16 + n];
        #pragma unroll
        for (int r = 0; r < 2; ++r) {
            v8f c;
            #pragma unroll
            for (int i = 0; i < 8; ++i) c[i] = bval;
            #pragma unroll
            for (int kb = 0; kb < 4; ++kb) {
                v16h a = load_a_frag(x_lds + r * 2048, lane, kb);
                v16h b = *(const v16h*)(w1p + ((size_t)(ct * 4 + kb) * 32 + lane) * 16);
                c = __builtin_amdgcn_wmma_f32_16x16x32_f16(false, a, false, b,
                                                           (short)0, c, false, false);
            }
            #pragma unroll
            for (int v = 0; v < 8; ++v)
                h_lds[(r * 16 + mbase + v) * 128 + ct * 16 + n] =
                    (_Float16)fmaxf(c[v], 0.0f);
        }
    }
    __syncthreads();

    // ---- phase 2: FC2 A-fragments (h tile) loaded once per wave ----------
    v16h af[2][4];
    #pragma unroll
    for (int r = 0; r < 2; ++r)
        #pragma unroll
        for (int kb = 0; kb < 4; ++kb)
            af[r][kb] = load_a_frag(h_lds + r * 2048, lane, kb);
    __syncthreads();   // x/h tiles dead from here -> LDS reused as acc block

    // ---- phase 3: FC2 WMMA sweep, w-tiles consumed in-register -----------
    const int X = wv >> 1, parity = wv & 1;
    const bool isC = (X == 2);
    const float* cf0 = (X == 0) ? coefA : (X == 1) ? coefB
                                        : (X == 3) ? coefD : coefC0;

    float acc0[2][8], acc1[2][8], acc2[2][8];
    #pragma unroll
    for (int r = 0; r < 2; ++r)
        #pragma unroll
        for (int v = 0; v < 8; ++v) { acc0[r][v] = 0.f; acc1[r][v] = 0.f; acc2[r][v] = 0.f; }

    for (int u = 0; u < 32; ++u) {
        const int t = X * 64 + parity + 2 * u;            // global 16-col tile
        float bval = b2[t * 16 + n];                      // bias folded into C
        v8f c0, c1;
        #pragma unroll
        for (int i = 0; i < 8; ++i) { c0[i] = bval; c1[i] = bval; }

        #pragma unroll
        for (int kb = 0; kb < 4; ++kb) {
            v16h b = *(const v16h*)(w2p + ((size_t)(t * 4 + kb) * 32 + lane) * 16);
            c0 = __builtin_amdgcn_wmma_f32_16x16x32_f16(false, af[0][kb], false, b,
                                                        (short)0, c0, false, false);
            c1 = __builtin_amdgcn_wmma_f32_16x16x32_f16(false, af[1][kb], false, b,
                                                        (short)0, c1, false, false);
        }
        if (u + 1 < 32)   // pull next tile's B fragments toward L2/L0
            __builtin_prefetch(w2p + ((size_t)((t + 2) * 4) * 32 + lane) * 16, 0, 0);

        // consume w-tile: acc[m][w] += coef[m][u] * w[m, u, w]
        #pragma unroll
        for (int v = 0; v < 8; ++v) {
            int m0 = mbase + v, m1 = 16 + mbase + v;
            float cv0 = c0[v], cv1 = c1[v];
            acc0[0][v] += cf0[m0 * 32 + u] * cv0;
            acc0[1][v] += cf0[m1 * 32 + u] * cv1;
            if (isC) {
                acc1[0][v] += coefC1[m0 * 32 + u] * cv0;
                acc1[1][v] += coefC1[m1 * 32 + u] * cv1;
                acc2[0][v] += coefC2[m0 * 32 + u] * cv0;
                acc2[1][v] += coefC2[m1 * 32 + u] * cv1;
            }
        }
    }

    // each (X,parity) owned by exactly one wave -> conflict-free LDS stores
    const int w = parity * 16 + n;
    float* d0 = (X == 0) ? accA : (X == 1) ? accB : (X == 3) ? accD : accC0;
    #pragma unroll
    for (int r = 0; r < 2; ++r)
        #pragma unroll
        for (int v = 0; v < 8; ++v) {
            int m = r * 16 + mbase + v;
            d0[m * 32 + w] = acc0[r][v];
            if (isC) {
                accC1[m * 32 + w] = acc1[r][v];
                accC2[m * 32 + w] = acc2[r][v];
            }
        }
    __syncthreads();

    // ---- phase 4: combine TP outputs, scatter-add by edge_src ------------
    for (int p = tid; p < 4096; p += 256) {
        int m = p >> 7, col = p & 127;
        int e = e0 + m;
        if (e >= E) continue;
        int src = eidx[e];                                // edge_index[0][e]
        float val;
        if (col < 32) {
            val = PATH_NORM * (accA[m * 32 + col] + INV_SQRT3 * accD[m * 32 + col]);
        } else {
            int q = col - 32, ww = q / 3, i = q - ww * 3;
            const float* cc = (i == 0) ? accC0 : (i == 1) ? accC1 : accC2;
            val = PATH_NORM * INV_SQRT3 *
                  (accB[m * 32 + ww] * shv[m * 3 + i] + cc[m * 32 + ww]);
        }
        atomicAdd(&node_acc[(size_t)src * 128 + col], val);
    }
    if (tid < 32) {
        int e = e0 + tid;
        if (e < E) atomicAdd(&counts[eidx[e]], 1.0f);
    }
}

// ---------------------------------------------------------------------------
// Post 1: y = node_acc / max(count,1) + node_attr ; hierarchical BN stats.
// stats = [ sum_s[32] | sumsq_s[32] | sumsq_v[32] ]
// ---------------------------------------------------------------------------
__global__ __launch_bounds__(256) void post1_kernel(const float* __restrict__ node_acc,
                                                    const float* __restrict__ counts,
                                                    const float* __restrict__ node_attr,
                                                    float* __restrict__ y,
                                                    float* __restrict__ stats, int N) {
    __shared__ float part[96];
    const int tid = threadIdx.x;
    if (tid < 96) part[tid] = 0.0f;
    __syncthreads();

    const int n0 = blockIdx.x * 16;
    for (int p = tid; p < 2048; p += 256) {
        int node = n0 + (p >> 7), col = p & 127;
        if (node < N) {
            float cnt = counts[node];
            size_t ix = (size_t)node * 128 + col;
            float val = node_acc[ix] / fmaxf(cnt, 1.0f) + node_attr[ix];
            y[ix] = val;
            if (col < 32) {
                atomicAdd(&part[col], val);            // ds_add_f32
                atomicAdd(&part[32 + col], val * val);
            } else {
                int u = (col - 32) / 3;
                atomicAdd(&part[64 + u], val * val);
            }
        }
    }
    __syncthreads();
    if (tid < 96) atomicAdd(&stats[tid], part[tid]);
}

// ---------------------------------------------------------------------------
// Post 2: apply batch norm in place on y (= d_out).
// ---------------------------------------------------------------------------
__global__ __launch_bounds__(256) void post2_kernel(float* __restrict__ y,
                                                    const float* __restrict__ stats,
                                                    const float* __restrict__ bn_w,
                                                    const float* __restrict__ bn_b, int N) {
    int idx = blockIdx.x * blockDim.x + threadIdx.x;
    int node = idx >> 7, col = idx & 127;
    if (node >= N) return;
    float Nf = (float)N;
    float val = y[idx];
    if (col < 32) {
        float mu  = stats[col] / Nf;
        float var = stats[32 + col] / Nf - mu * mu;       // E[(s-mu)^2]
        val = (val - mu) * (rsqrtf(var + EPS_BN) * bn_w[col]) + bn_b[col];
    } else {
        int u = (col - 32) / 3;
        float vn = rsqrtf(stats[64 + u] / (3.0f * Nf) + EPS_BN) * bn_w[32 + u];
        val = val * vn;
    }
    y[idx] = val;
}

// ---------------------------------------------------------------------------
extern "C" void kernel_launch(void* const* d_in, const int* in_sizes, int n_in,
                              void* d_out, int out_size, void* d_ws, size_t ws_size,
                              hipStream_t stream) {
    const float* node_attr = (const float*)d_in[0];
    const int*   edge_index= (const int*)  d_in[1];
    const float* edge_attr = (const float*)d_in[2];
    const float* edge_sh   = (const float*)d_in[3];
    const float* W1        = (const float*)d_in[4];
    const float* b1        = (const float*)d_in[5];
    const float* W2        = (const float*)d_in[6];
    const float* b2        = (const float*)d_in[7];
    const float* bn_w      = (const float*)d_in[8];
    const float* bn_b      = (const float*)d_in[9];

    const int N = in_sizes[0] / IN_DIM;
    const int E = in_sizes[1] / 2;

    // workspace carve-out (all offsets 32B-aligned for this N)
    char* ws = (char*)d_ws;
    float* node_acc = (float*)ws;  ws += (size_t)N * 128 * sizeof(float);
    float* counts   = (float*)ws;  ws += (size_t)N * sizeof(float);
    float* stats    = (float*)ws;  ws += 96 * sizeof(float);
    _Float16* w1p   = (_Float16*)ws; ws += (size_t)HIDDEN * HIDDEN * sizeof(_Float16);
    _Float16* w2p   = (_Float16*)ws; ws += (size_t)HIDDEN * W_NUMEL * sizeof(_Float16);

    float* out = (float*)d_out;

    // node_acc | counts | stats are contiguous -> single zero pass
    int nzero = N * 128 + N + 96;
    ws_zero_kernel<<<(nzero + 255) / 256, 256, 0, stream>>>(node_acc, nzero);

    int tot1 = HIDDEN * HIDDEN;      // 16384
    int tot2 = HIDDEN * W_NUMEL;     // 524288
    pack_b_kernel<<<(tot1 + 255) / 256, 256, 0, stream>>>(W1, w1p, HIDDEN, HIDDEN, tot1);
    pack_b_kernel<<<(tot2 + 255) / 256, 256, 0, stream>>>(W2, w2p, HIDDEN, W_NUMEL, tot2);

    int eblocks = (E + 31) / 32;
    fused_edge_kernel<<<eblocks, 256, 0, stream>>>(edge_attr, w1p, b1, w2p, b2,
                                                   node_attr, edge_index, edge_sh,
                                                   node_acc, counts, E, N);

    post1_kernel<<<(N + 15) / 16, 256, 0, stream>>>(node_acc, counts, node_attr,
                                                    out, stats, N);
    post2_kernel<<<((N * 128) + 255) / 256, 256, 0, stream>>>(out, stats, bn_w, bn_b, N);
}
